// GATHead_6700148981953
// MI455X (gfx1250) — matrix operand order repaired and checked by reading, not compiled, not run on hardware
//
#include <hip/hip_runtime.h>
#include <hip/hip_bf16.h>
#include <stdint.h>

// ---------------------------------------------------------------------------
// Types for CDNA5 WMMA / TDM
// ---------------------------------------------------------------------------
typedef __attribute__((ext_vector_type(16))) __bf16       v16bf;
typedef __attribute__((ext_vector_type(8)))  float        v8f;
typedef __attribute__((ext_vector_type(4)))  unsigned int u32x4;
typedef __attribute__((ext_vector_type(8)))  int          i32x8;
typedef __attribute__((ext_vector_type(4)))  int          i32x4;

#if __has_builtin(__builtin_amdgcn_tensor_load_to_lds)
#define HAVE_TDM 1
#else
#define HAVE_TDM 0
#endif

#define NEG_SLOPE 0.2f

__device__ __forceinline__ unsigned short f32_to_bf16(float f) {
  unsigned u = __float_as_uint(f);
  unsigned r = (u + 0x7FFFu + ((u >> 16) & 1u)) >> 16;   // round-to-nearest-even
  return (unsigned short)r;
}

__device__ __forceinline__ void atomic_add_f32(float* p, float v) {
  __hip_atomic_fetch_add(p, v, __ATOMIC_RELAXED, __HIP_MEMORY_SCOPE_AGENT);
}

// ---------------------------------------------------------------------------
// TDM 2D tile load descriptor (D# groups per CDNA5 ISA 08_async_tensor §8.3-8.6)
// data_size code: 0=1B 1=2B 2=4B 3=8B. dims/strides in data_size units.
// dim0_rem/dim1_rem are the remaining tensor extents from the tile start
// (rows beyond dim1_rem are zero-filled -> handles ragged M edge tiles).
// 6-arg builtin on this toolchain:
//   (u32x4 g0, i32x8 g1, i32x4 g2, i32x4 g3, i32x8 pad/VADDR4, i32 cpol)
// ---------------------------------------------------------------------------
#if HAVE_TDM
__device__ __forceinline__ void tdm_load_2d(unsigned lds_off, const void* gptr,
                                            unsigned dim0_rem, unsigned dim1_rem,
                                            unsigned tile0, unsigned tile1,
                                            unsigned stride0, unsigned dsz) {
  unsigned long long ga = (unsigned long long)(uintptr_t)gptr;
  u32x4 g0;
  g0.x = 1u;                                              // count=1, user mode
  g0.y = lds_off;                                         // lds_addr [63:32]
  g0.z = (unsigned)(ga & 0xFFFFFFFFull);                  // global_addr[31:0]
  g0.w = (unsigned)((ga >> 32) & 0x1FFFFFFull) | (2u << 30); // addr[56:32] | type=2
  i32x8 g1;
  g1[0] = (int)(dsz << 16);                               // wg_mask=0, data_size
  g1[1] = (int)((dim0_rem & 0xFFFFu) << 16);              // tensor_dim0[15:0]
  g1[2] = (int)((dim0_rem >> 16) | ((dim1_rem & 0xFFFFu) << 16));
  g1[3] = (int)((dim1_rem >> 16) | (tile0 << 16));        // tile_dim0
  g1[4] = (int)(tile1 & 0xFFFFu);                         // tile_dim1 (tile_dim2=0)
  g1[5] = (int)stride0;                                   // tensor_dim0_stride lo
  g1[6] = 0;
  g1[7] = 0;
  i32x4 gz; gz[0] = 0; gz[1] = 0; gz[2] = 0; gz[3] = 0;   // groups 2/3: 2D tensor
  i32x8 gz8;
  gz8[0] = 0; gz8[1] = 0; gz8[2] = 0; gz8[3] = 0;
  gz8[4] = 0; gz8[5] = 0; gz8[6] = 0; gz8[7] = 0;         // unused VADDR4 slot
  __builtin_amdgcn_tensor_load_to_lds(g0, g1, gz, gz, gz8, 0);
}
#endif

// ---------------------------------------------------------------------------
// bf16 WMMA GEMM:  C[M,N] f32 = A[M,K] @ Bt[N,K]^T  (A row-major, B given
// TRANSPOSED so both LDS tiles are K-contiguous -> all fragment loads are
// 16B ds_load_b128, no u16 gathers).
// 256 threads = 8 waves; block tile 128x64; wave tile 32x32 (2x2 WMMA frags);
// K-step 32.  Double-buffered LDS, tiles prefetched by the Tensor Data Mover
// so the async DMA for step k+1 overlaps the WMMA work of step k.
// ---------------------------------------------------------------------------
#define BM 128
#define BN 64
#define BK 32

__global__ __launch_bounds__(256) void gemm_bf16_wmma(
    const unsigned short* __restrict__ A,    // [M,K] bf16
    const unsigned short* __restrict__ Bt,   // [N,K] bf16 (transposed B)
    float* __restrict__ C, int M, int N, int K) {
  __shared__ unsigned short Abuf[2][BM * BK];   // 2 x 8 KB
  __shared__ unsigned short Bbuf[2][BN * BK];   // 2 x 4 KB

  const int tid  = threadIdx.x;
  const int lane = tid & 31;
  const int wave = tid >> 5;
  const int half = lane >> 4;
  const int l16  = lane & 15;

  const int m0 = blockIdx.x * BM;
  const int n0 = blockIdx.y * BN;
  const int mw = (wave >> 1) * 32;           // wave M offset (0..96)
  const int nw = (wave & 1) * 32;            // wave N offset (0..32)

  v8f zero = {0.f, 0.f, 0.f, 0.f, 0.f, 0.f, 0.f, 0.f};
  v8f acc[2][2];
#pragma unroll
  for (int i = 0; i < 2; ++i)
#pragma unroll
    for (int j = 0; j < 2; ++j) acc[i][j] = zero;

  const int steps = K / BK;

#if HAVE_TDM
  unsigned ldsA[2], ldsB[2];
  ldsA[0] = (unsigned)(uintptr_t)(&Abuf[0][0]);   // flat LDS addr low 32b = offset
  ldsA[1] = (unsigned)(uintptr_t)(&Abuf[1][0]);
  ldsB[0] = (unsigned)(uintptr_t)(&Bbuf[0][0]);
  ldsB[1] = (unsigned)(uintptr_t)(&Bbuf[1][0]);
  auto issue = [&](int buf, int it) {
    const int k0 = it * BK;
    tdm_load_2d(ldsA[buf], A + (size_t)m0 * K + k0,
                (unsigned)(K - k0), (unsigned)(M - m0), BK, BM, (unsigned)K, 1u);
    tdm_load_2d(ldsB[buf], Bt + (size_t)n0 * K + k0,
                (unsigned)(K - k0), (unsigned)(N - n0), BK, BN, (unsigned)K, 1u);
  };
  if (wave == 0) issue(0, 0);
#endif

  for (int it = 0; it < steps; ++it) {
    const int cur = it & 1;
#if HAVE_TDM
    if (wave == 0) __builtin_amdgcn_s_wait_tensorcnt(0);  // tile `it` landed
    __syncthreads();                                      // visible to all waves
    if (wave == 0 && it + 1 < steps) issue(1 - cur, it + 1);  // prefetch overlap
#else
    const int k0 = it * BK;
    for (int l = tid; l < BM * BK; l += 256) {
      int r = l / BK, c = l - r * BK;
      int gm = m0 + r;
      Abuf[cur][l] = (gm < M) ? A[(size_t)gm * K + k0 + c] : (unsigned short)0;
    }
    for (int l = tid; l < BN * BK; l += 256) {
      int r = l / BK, c = l - r * BK;
      Bbuf[cur][l] = Bt[(size_t)(n0 + r) * K + k0 + c];
    }
    __syncthreads();
#endif

    // A fragment (16x32 bf16): lane l16 = row M; dwords [4*half..4*half+3] and
    // [8+4*half..8+4*half+3] -> two contiguous 16B LDS reads.
    v16bf a[2], b[2];
#pragma unroll
    for (int i = 0; i < 2; ++i) {
      const unsigned* Arow = (const unsigned*)(&Abuf[cur][(mw + i * 16 + l16) * BK]);
      union { unsigned u[8]; v16bf v; } af;
#pragma unroll
      for (int r = 0; r < 4; ++r) af.u[r] = Arow[4 * half + r];
#pragma unroll
      for (int r = 0; r < 4; ++r) af.u[4 + r] = Arow[8 + 4 * half + r];
      a[i] = af.v;
    }
    // B fragment (32x16 bf16): lane l16 = col N; dwords [4*half..4*half+3] and
    // [8+4*half..8+4*half+3] -> two contiguous 16B LDS reads (same K order as A).
#pragma unroll
    for (int j = 0; j < 2; ++j) {
      const unsigned* Brow = (const unsigned*)(&Bbuf[cur][(nw + j * 16 + l16) * BK]);
      union { unsigned u[8]; v16bf v; } bfv;
#pragma unroll
      for (int r = 0; r < 4; ++r) bfv.u[r] = Brow[4 * half + r];
#pragma unroll
      for (int r = 0; r < 4; ++r) bfv.u[4 + r] = Brow[8 + 4 * half + r];
      b[j] = bfv.v;
    }
#pragma unroll
    for (int i = 0; i < 2; ++i)
#pragma unroll
      for (int j = 0; j < 2; ++j)
        acc[i][j] = __builtin_amdgcn_wmma_f32_16x16x32_bf16(
            false, a[i], false, b[j], (short)0, acc[i][j], false, false);

    __syncthreads();   // all waves done with `cur` before TDM reuses it
  }

  // C/D layout: lane l16 = col N, VGPR r -> row M = r + 8*half
#pragma unroll
  for (int i = 0; i < 2; ++i) {
    int mt = m0 + mw + i * 16;
#pragma unroll
    for (int j = 0; j < 2; ++j) {
      int n = n0 + nw + j * 16 + l16;
#pragma unroll
      for (int r = 0; r < 8; ++r) {
        int m = mt + r + 8 * half;
        if (m < M) C[(size_t)m * N + n] = acc[i][j][r];
      }
    }
  }
}

// ---------------------------------------------------------------------------
// fp32 -> bf16 cast
// ---------------------------------------------------------------------------
__global__ void cast_bf16(const float* __restrict__ in, unsigned short* __restrict__ out, int n) {
  int i = blockIdx.x * blockDim.x + threadIdx.x;
  if (i < n) out[i] = f32_to_bf16(in[i]);
}

// ---------------------------------------------------------------------------
// fp32 W[K,N] -> bf16 Wt[N,K] (one-time, tiny weight matrices)
// ---------------------------------------------------------------------------
__global__ void cast_transpose_bf16(const float* __restrict__ W, unsigned short* __restrict__ Wt,
                                    int K, int N) {
  int i = blockIdx.x * blockDim.x + threadIdx.x;
  if (i >= K * N) return;
  int k = i / N, n = i - k * N;          // coalesced read along N
  Wt[(size_t)n * K + k] = f32_to_bf16(W[i]);
}

// ---------------------------------------------------------------------------
// a_s[n,h] = <xl[n,h,:], att_src[h,:]>, a_d likewise.  One wave per (n,h).
// ---------------------------------------------------------------------------
__global__ __launch_bounds__(256) void att_scores(
    const float* __restrict__ xl, const float* __restrict__ att_s,
    const float* __restrict__ att_d, float* __restrict__ as_, float* __restrict__ ad_,
    int N, int H, int C) {
  int wave = blockIdx.x * (blockDim.x >> 5) + (threadIdx.x >> 5);
  int lane = threadIdx.x & 31;
  if (wave >= N * H) return;
  int n = wave / H, h = wave - n * H;
  const float* row = xl + (size_t)n * H * C + (size_t)h * C;
  const float* sv = att_s + h * C;
  const float* dv = att_d + h * C;
  float ss = 0.f, sd = 0.f;
  for (int c = lane; c < C; c += 32) {
    float v = row[c];
    ss += v * sv[c];
    sd += v * dv[c];
  }
#pragma unroll
  for (int o = 16; o > 0; o >>= 1) {          // wave32 reduction
    ss += __shfl_xor(ss, o, 32);
    sd += __shfl_xor(sd, o, 32);
  }
  if (lane == 0) { as_[wave] = ss; ad_[wave] = sd; }
}

// ---------------------------------------------------------------------------
// Edge pass 1: e = leaky_relu(a_s[src]+a_d[dst]); ex = exp(e); denom[dst] += ex
// (softmax shift-invariance => segment_max elided; |e| is O(1) here)
// ---------------------------------------------------------------------------
__global__ void edge_softmax(const int* __restrict__ ei, const float* __restrict__ as_,
                             const float* __restrict__ ad_, float* __restrict__ ex,
                             float* __restrict__ denom, int E, int Nn, int H) {
  int t = blockIdx.x * blockDim.x + threadIdx.x;
  int total = (E + Nn) * H;
  if (t >= total) return;
  int idx = t / H, h = t - idx * H;
  int s, d;
  if (idx < E) { s = ei[idx]; d = ei[E + idx]; } else { s = d = idx - E; }  // self-loops
  float e = as_[s * H + h] + ad_[d * H + h];
  e = (e > 0.f) ? e : NEG_SLOPE * e;
  float x = __expf(e);
  ex[t] = x;
  atomic_add_f32(&denom[d * H + h], x);
}

// ---------------------------------------------------------------------------
// Edge pass 2: out[dst,h,:] += alpha * xl[src,h,:].  One wave per (edge,head);
// coalesced 32-lane channel strides; scatter via L2-resident f32 atomics.
// ---------------------------------------------------------------------------
__global__ __launch_bounds__(256) void edge_message(
    const int* __restrict__ ei, const float* __restrict__ xl, const float* __restrict__ ex,
    const float* __restrict__ denom, float* __restrict__ out, int E, int Nn, int H, int C) {
  int wave = blockIdx.x * (blockDim.x >> 5) + (threadIdx.x >> 5);
  int lane = threadIdx.x & 31;
  int total = (E + Nn) * H;
  if (wave >= total) return;
  int idx = wave / H, h = wave - idx * H;
  int s, d;
  if (idx < E) { s = ei[idx]; d = ei[E + idx]; } else { s = d = idx - E; }
  float alpha = ex[wave] / denom[d * H + h];
  const float* srow = xl + (size_t)s * H * C + (size_t)h * C;
  float* drow = out + (size_t)d * H * C + (size_t)h * C;
  for (int c = lane; c < C; c += 32) atomic_add_f32(&drow[c], srow[c] * alpha);
}

// ---------------------------------------------------------------------------
// Layer-1 epilogue: h1 = relu(acc + bias), written directly as bf16 (it only
// feeds the next WMMA GEMM).
// ---------------------------------------------------------------------------
__global__ void relu_bias_bf16(const float* __restrict__ acc, const float* __restrict__ bias,
                               unsigned short* __restrict__ hbf, int n, int HC) {
  int i = blockIdx.x * blockDim.x + threadIdx.x;
  if (i >= n) return;
  float v = acc[i] + bias[i % HC];
  v = (v > 0.f) ? v : 0.f;
  hbf[i] = f32_to_bf16(v);
}

// ---------------------------------------------------------------------------
// Layer-2 epilogue: h2 = relu(acc + bias); column sums for global mean pool.
// ---------------------------------------------------------------------------
__global__ void relu_bias_colsum(const float* __restrict__ acc, const float* __restrict__ bias,
                                 float* __restrict__ zsum, int n, int C) {
  int i = blockIdx.x * blockDim.x + threadIdx.x;
  if (i >= n) return;
  int c = i % C;
  float v = acc[i] + bias[c];
  v = (v > 0.f) ? v : 0.f;
  atomic_add_f32(&zsum[c], v);
}

// ---------------------------------------------------------------------------
// Final 1x256 @ 256x10 head.
// ---------------------------------------------------------------------------
__global__ void final_head(const float* __restrict__ zsum, const float* __restrict__ Wc,
                           const float* __restrict__ bc, float* __restrict__ out,
                           int N, int C, int ncls) {
  int j = threadIdx.x;
  if (j >= ncls) return;
  float inv = 1.0f / (float)N;
  float s = bc[j];
  for (int c = 0; c < C; ++c) s += (zsum[c] * inv) * Wc[c * ncls + j];
  out[j] = s;
}

// ---------------------------------------------------------------------------
// Host-side orchestration (graph-capture safe: only async ops on `stream`).
// ---------------------------------------------------------------------------
extern "C" void kernel_launch(void* const* d_in, const int* in_sizes, int n_in,
                              void* d_out, int out_size, void* d_ws, size_t ws_size,
                              hipStream_t stream) {
  (void)n_in; (void)out_size; (void)ws_size;
  const float* x    = (const float*)d_in[0];
  const int*   ei   = (const int*)d_in[1];      // edge_index [2,E] (int32 under default JAX)
  const float* W1   = (const float*)d_in[2];
  const float* as1w = (const float*)d_in[3];
  const float* ad1w = (const float*)d_in[4];
  const float* b1   = (const float*)d_in[5];
  const float* W2   = (const float*)d_in[6];
  const float* as2w = (const float*)d_in[7];
  const float* ad2w = (const float*)d_in[8];
  const float* b2   = (const float*)d_in[9];
  const float* Wc   = (const float*)d_in[10];
  const float* bc   = (const float*)d_in[11];

  const int D = 128, H1 = 4, C = 256, NCLS = 10;
  const int N  = in_sizes[0] / D;       // 10000
  const int E  = in_sizes[1] / 2;       // 160000
  const int HC = H1 * C;                // 1024
  const int TE = E + N;                 // edges + self-loops

  // Workspace layout (layer-2 buffers alias layer-1 buffers that are dead by then).
  char* base = (char*)d_ws;
  size_t off = 0;
  auto alloc = [&](size_t bytes) -> char* {
    char* p = base + off;
    off += (bytes + 255) & ~(size_t)255;
    return p;
  };
  unsigned short* Xbf  = (unsigned short*)alloc((size_t)N * D * 2);
  unsigned short* W1t  = (unsigned short*)alloc((size_t)D * HC * 2);   // [HC,D] bf16
  unsigned short* W2t  = (unsigned short*)alloc((size_t)HC * C * 2);   // [C,HC] bf16
  float* xl1  = (float*)alloc((size_t)N * HC * 4);    // reused as xl2
  float* out1 = (float*)alloc((size_t)N * HC * 4);    // reused as out2
  unsigned short* h1bf = (unsigned short*)alloc((size_t)N * HC * 2);
  float* as1  = (float*)alloc((size_t)N * H1 * 4);    // reused as as2
  float* ad1  = (float*)alloc((size_t)N * H1 * 4);    // reused as ad2
  float* ex1  = (float*)alloc((size_t)TE * H1 * 4);   // reused as ex2
  float* dn1  = (float*)alloc((size_t)N * H1 * 4);    // reused as dn2
  float* zsum = (float*)alloc((size_t)C * 4);
  float* xl2 = xl1; float* out2 = out1; float* as2 = as1; float* ad2 = ad1;
  float* ex2 = ex1; float* dn2 = dn1;

  // bf16 operands for the WMMA GEMMs (weights pre-transposed so GEMM B-tiles
  // are K-contiguous -> TDM tiles land in the fragment-friendly LDS layout)
  cast_bf16<<<dim3((N * D + 255) / 256), dim3(256), 0, stream>>>(x, Xbf, N * D);
  cast_transpose_bf16<<<dim3((D * HC + 255) / 256), dim3(256), 0, stream>>>(W1, W1t, D, HC);
  cast_transpose_bf16<<<dim3((HC * C + 255) / 256), dim3(256), 0, stream>>>(W2, W2t, HC, C);

  // ---- Layer 1 (H=4, C=256) ----
  (void)hipMemsetAsync(dn1, 0, (size_t)N * H1 * 4, stream);
  (void)hipMemsetAsync(out1, 0, (size_t)N * HC * 4, stream);
  gemm_bf16_wmma<<<dim3((N + BM - 1) / BM, HC / BN), dim3(256), 0, stream>>>(
      Xbf, W1t, xl1, N, HC, D);
  att_scores<<<dim3((N * H1 + 7) / 8), dim3(256), 0, stream>>>(
      xl1, as1w, ad1w, as1, ad1, N, H1, C);
  edge_softmax<<<dim3((TE * H1 + 255) / 256), dim3(256), 0, stream>>>(
      ei, as1, ad1, ex1, dn1, E, N, H1);
  edge_message<<<dim3((TE * H1 + 7) / 8), dim3(256), 0, stream>>>(
      ei, xl1, ex1, dn1, out1, E, N, H1, C);
  relu_bias_bf16<<<dim3((N * HC + 255) / 256), dim3(256), 0, stream>>>(
      out1, b1, h1bf, N * HC, HC);

  // ---- Layer 2 (H=1, C=256) ----
  (void)hipMemsetAsync(dn2, 0, (size_t)N * 4, stream);
  (void)hipMemsetAsync(out2, 0, (size_t)N * C * 4, stream);
  (void)hipMemsetAsync(zsum, 0, (size_t)C * 4, stream);
  gemm_bf16_wmma<<<dim3((N + BM - 1) / BM, C / BN), dim3(256), 0, stream>>>(
      h1bf, W2t, xl2, N, C, HC);
  att_scores<<<dim3((N + 7) / 8), dim3(256), 0, stream>>>(
      xl2, as2w, ad2w, as2, ad2, N, 1, C);
  edge_softmax<<<dim3((TE + 255) / 256), dim3(256), 0, stream>>>(
      ei, as2, ad2, ex2, dn2, E, N, 1);
  edge_message<<<dim3((TE + 7) / 8), dim3(256), 0, stream>>>(
      ei, xl2, ex2, dn2, out2, E, N, 1, C);
  relu_bias_colsum<<<dim3((N * C + 255) / 256), dim3(256), 0, stream>>>(
      out2, b2, zsum, N * C, C);

  // ---- Mean pool + classifier head ----
  final_head<<<dim3(1), dim3(64), 0, stream>>>(zsum, Wc, bc, (float*)d_out, N, C, NCLS);
}
